// LinearAttention_27444841021534
// MI455X (gfx1250) — matrix operand order repaired and checked by reading, not compiled, not run on hardware
//
#include <hip/hip_runtime.h>

// ---------------------------------------------------------------------------
// Multi-head attention block for MI455X (gfx1250, wave32, WMMA bf16 16x16x32)
//   1) qkv = x @ w_qkv + b_qkv   -> q [bh][n][d], kT [bh][d][n], v [bh][n][d] (bf16)
//   2) flash attention per (bh, 16-row query tile)  -> ao [row][1024] (bf16)
//   3) out = ao @ w_proj + b_proj (fp32)
// Double-buffered LDS staging in both GEMMs so global loads for tile k+1
// overlap the WMMAs of tile k; uniform (scalar-branch) epilogue scatter.
// ---------------------------------------------------------------------------

typedef __attribute__((ext_vector_type(16))) __bf16 v16bf;
typedef __attribute__((ext_vector_type(8)))  __bf16 v8bf;
typedef __attribute__((ext_vector_type(8)))  float  v8f;

#define EMBED 1024
#define NH    16
#define HD    64
#define SEQ   2048
#define BATCH 2
#define ROWS  (BATCH * SEQ)      // 4096
#define NQKV  (3 * EMBED)        // 3072
#define ATT_SCALE 0.125f         // HD^-0.5

__device__ __forceinline__ v8f v8f_zero() {
  v8f z;
#pragma unroll
  for (int i = 0; i < 8; ++i) z[i] = 0.0f;
  return z;
}

__device__ __forceinline__ v8f wmma_bf16(v16bf a, v16bf b, v8f c) {
  // (neg_a, A, neg_b, B, c_mod, C, reuse_a, reuse_b)
  return __builtin_amdgcn_wmma_f32_16x16x32_bf16(false, a, false, b, (short)0, c,
                                                 false, false);
}

// A-fragment (16x32 bf16): lane holds row (lane&15), K in {klo..klo+7, klo+16..klo+23}
__device__ __forceinline__ v16bf load_a_frag(const __bf16* base, int stride, int lane) {
  const int r   = lane & 15;
  const int klo = (lane & 16) ? 8 : 0;
  const __bf16* p = base + (size_t)r * stride + klo;
  union { v16bf v; v8bf h[2]; } u;
  u.h[0] = *(const v8bf*)(p);
  u.h[1] = *(const v8bf*)(p + 16);
  return u.v;
}

// B-fragment (32x16 bf16): lane holds col N=(lane&15), K in {klo..klo+7, klo+16..klo+23}
__device__ __forceinline__ v16bf load_b_frag(const __bf16* base, int stride, int lane) {
  const int n   = lane & 15;
  const int klo = (lane & 16) ? 8 : 0;
  v16bf v;
#pragma unroll
  for (int s = 0; s < 8; ++s) {
    v[s]     = base[(size_t)(klo + s) * stride + n];
    v[s + 8] = base[(size_t)(klo + 16 + s) * stride + n];
  }
  return v;
}

__device__ __forceinline__ void store_bf16x4(__bf16* d, float4 v) {
  d[0] = (__bf16)v.x; d[1] = (__bf16)v.y; d[2] = (__bf16)v.z; d[3] = (__bf16)v.w;
}

// ---------------------------------------------------------------------------
// Kernel 1: QKV projection.  out tile 128x128 per workgroup, 8 waves x (32x64).
// ---------------------------------------------------------------------------
__global__ __launch_bounds__(256) void qkv_gemm_kernel(
    const float* __restrict__ x, const float* __restrict__ w,
    const float* __restrict__ bias,
    __bf16* __restrict__ qb, __bf16* __restrict__ ktb, __bf16* __restrict__ vb)
{
  __shared__ __bf16 As[2][128 * 40];   // x tile [m][k], padded stride 40
  __shared__ __bf16 Bs[2][32 * 136];   // w tile [k][n], padded stride 136

  const int t = threadIdx.x;
  const int wave = t >> 5, lane = t & 31;
  const int waveM = wave & 3, waveN = wave >> 2;
  const int m0 = blockIdx.x * 128;
  const int n0 = blockIdx.y * 128;

  v8f acc[2][4];
#pragma unroll
  for (int i = 0; i < 2; ++i)
#pragma unroll
    for (int j = 0; j < 4; ++j) acc[i][j] = v8f_zero();

  const int arow = t >> 3, acol = (t & 7) * 4;   // A staging: 32 rows x 8 col-groups
  const int bcol = (t & 31) * 4, brow = t >> 5;  // B staging: 8 k-rows x 32 col-groups

  float4 ar[4], br[4];
  auto load_tiles = [&](int k0) {
#pragma unroll
    for (int rr = 0; rr < 4; ++rr)
      ar[rr] = *(const float4*)(x + (size_t)(m0 + arow + rr * 32) * EMBED + k0 + acol);
#pragma unroll
    for (int rr = 0; rr < 4; ++rr)
      br[rr] = *(const float4*)(w + (size_t)(k0 + brow + rr * 8) * NQKV + n0 + bcol);
  };
  auto stage_tiles = [&](int buf) {
#pragma unroll
    for (int rr = 0; rr < 4; ++rr)
      store_bf16x4(&As[buf][(arow + rr * 32) * 40 + acol], ar[rr]);
#pragma unroll
    for (int rr = 0; rr < 4; ++rr)
      store_bf16x4(&Bs[buf][(brow + rr * 8) * 136 + bcol], br[rr]);
  };

  load_tiles(0);
  stage_tiles(0);
  __syncthreads();

  const int NK = EMBED / 32;
  for (int kt = 0; kt < NK; ++kt) {
    const int cur = kt & 1;
    if (kt + 1 < NK) load_tiles((kt + 1) * 32);   // overlaps WMMAs below

    v16bf af[2], bfr[4];
#pragma unroll
    for (int i = 0; i < 2; ++i)
      af[i] = load_a_frag(&As[cur][(waveM * 32 + i * 16) * 40], 40, lane);
#pragma unroll
    for (int j = 0; j < 4; ++j)
      bfr[j] = load_b_frag(&Bs[cur][waveN * 64 + j * 16], 136, lane);
#pragma unroll
    for (int i = 0; i < 2; ++i)
#pragma unroll
      for (int j = 0; j < 4; ++j) acc[i][j] = wmma_bf16(af[i], bfr[j], acc[i][j]);

    if (kt + 1 < NK) stage_tiles(cur ^ 1);
    __syncthreads();
  }

  // Epilogue. The 128-col tile lies entirely in one of q/k/v (uniform), and each
  // wave's 64-col slice lies in one head (uniform). Bias hoisted to 4 regs.
  const int which = n0 >> 10;                    // 0=q 1=k 2=v, block-uniform
  const int h     = (n0 & 1023) >> 6;            // head of cols [n0, n0+64)
  const int hh    = h + waveN;                   // waveN*64 advances one head
  float bias_r[4];
#pragma unroll
  for (int j = 0; j < 4; ++j)
    bias_r[j] = bias[n0 + waveN * 64 + j * 16 + (lane & 15)];

#pragma unroll
  for (int i = 0; i < 2; ++i)
#pragma unroll
    for (int r = 0; r < 8; ++r) {
      const int mrow = m0 + waveM * 32 + i * 16 + ((lane < 16) ? r : (r + 8));
      const int b = mrow >> 11, n = mrow & 2047;
      const size_t bh = (size_t)(b * NH + hh);
#pragma unroll
      for (int j = 0; j < 4; ++j) {
        const int d = j * 16 + (lane & 15);
        const __bf16 bv = (__bf16)(acc[i][j][r] + bias_r[j]);
        if (which == 0)      qb [(bh * SEQ + n) * HD + d] = bv;
        else if (which == 1) ktb[(bh * HD + d) * SEQ + n] = bv;   // transposed
        else                 vb [(bh * SEQ + n) * HD + d] = bv;
      }
    }
}

// ---------------------------------------------------------------------------
// Kernel 2: flash attention. block = 8 waves; each wave owns 16 query rows and
// streams all 2048 keys in chunks of 32 with online softmax.
// ---------------------------------------------------------------------------
__global__ __launch_bounds__(256) void attn_kernel(
    const __bf16* __restrict__ qb, const __bf16* __restrict__ ktb,
    const __bf16* __restrict__ vb, __bf16* __restrict__ ao)
{
  __shared__ __bf16 Ps[8][16 * 40];   // per-wave P tile (16 rows x 32 keys, padded)
  const int t = threadIdx.x, wave = t >> 5, lane = t & 31;
  const int bh = blockIdx.x;                 // 0..31
  const int q0 = blockIdx.y * 128 + wave * 16;

  const __bf16* qbase = qb  + (size_t)bh * SEQ * HD;
  const __bf16* kbase = ktb + (size_t)bh * HD * SEQ;   // [d][n]
  const __bf16* vbase = vb  + (size_t)bh * SEQ * HD;   // [n][d]
  __bf16* pt = &Ps[wave][0];

  const v16bf qf0 = load_a_frag(qbase + (size_t)q0 * HD,      HD, lane); // d 0..31
  const v16bf qf1 = load_a_frag(qbase + (size_t)q0 * HD + 32, HD, lane); // d 32..63

  v8f o[4];
#pragma unroll
  for (int tl = 0; tl < 4; ++tl) o[tl] = v8f_zero();
  float mrow[8], lrow[8];
#pragma unroll
  for (int r = 0; r < 8; ++r) { mrow[r] = -3.0e38f; lrow[r] = 0.0f; }

  for (int j0 = 0; j0 < SEQ; j0 += 32) {
    // S = Q @ K^T  (B-fragments: rows = d, cols = keys; contiguous in kT layout)
    const v16bf kf00 = load_b_frag(kbase + (size_t)0  * SEQ + j0,      SEQ, lane);
    const v16bf kf01 = load_b_frag(kbase + (size_t)32 * SEQ + j0,      SEQ, lane);
    const v16bf kf10 = load_b_frag(kbase + (size_t)0  * SEQ + j0 + 16, SEQ, lane);
    const v16bf kf11 = load_b_frag(kbase + (size_t)32 * SEQ + j0 + 16, SEQ, lane);
    v8f s0 = v8f_zero(), s1 = v8f_zero();
    s0 = wmma_bf16(qf0, kf00, s0); s0 = wmma_bf16(qf1, kf01, s0);
    s1 = wmma_bf16(qf0, kf10, s1); s1 = wmma_bf16(qf1, kf11, s1);

    // Online softmax; row r stats replicated across the 16-lane half that owns it.
#pragma unroll
    for (int r = 0; r < 8; ++r) {
      const float a = s0[r] * ATT_SCALE;
      const float c = s1[r] * ATT_SCALE;
      float mx = fmaxf(a, c);
#pragma unroll
      for (int off = 1; off < 16; off <<= 1) mx = fmaxf(mx, __shfl_xor(mx, off, 32));
      const float mn = fmaxf(mrow[r], mx);
      const float sf = __expf(mrow[r] - mn);
      mrow[r] = mn;
      const float p0 = __expf(a - mn);
      const float p1 = __expf(c - mn);
      float ps = p0 + p1;
#pragma unroll
      for (int off = 1; off < 16; off <<= 1) ps += __shfl_xor(ps, off, 32);
      lrow[r] = lrow[r] * sf + ps;
#pragma unroll
      for (int tl = 0; tl < 4; ++tl) o[tl][r] *= sf;
      const int prow = (lane < 16) ? r : (r + 8);
      pt[prow * 40 + (lane & 15)]      = (__bf16)p0;
      pt[prow * 40 + 16 + (lane & 15)] = (__bf16)p1;
    }

    // O += P @ V   (P re-shaped via wave-private LDS into A-fragment layout)
    const v16bf pf = load_a_frag(pt, 40, lane);
#pragma unroll
    for (int tl = 0; tl < 4; ++tl) {
      const v16bf vf = load_b_frag(vbase + (size_t)j0 * HD + tl * 16, HD, lane);
      o[tl] = wmma_bf16(pf, vf, o[tl]);
    }
  }

  const int b = bh >> 4, h = bh & 15;
  float rinv[8];
#pragma unroll
  for (int r = 0; r < 8; ++r) rinv[r] = 1.0f / lrow[r];
#pragma unroll
  for (int tl = 0; tl < 4; ++tl)
#pragma unroll
    for (int r = 0; r < 8; ++r) {
      const int n   = q0 + ((lane < 16) ? r : (r + 8));
      const int col = h * HD + tl * 16 + (lane & 15);
      ao[((size_t)(b * SEQ + n)) * EMBED + col] = (__bf16)(o[tl][r] * rinv[r]);
    }
}

// ---------------------------------------------------------------------------
// Kernel 3: output projection. A read as bf16 fragments directly from global,
// B double-buffered through LDS; fp32 output + bias.
// ---------------------------------------------------------------------------
__global__ __launch_bounds__(256) void proj_gemm_kernel(
    const __bf16* __restrict__ ao, const float* __restrict__ w,
    const float* __restrict__ bias, float* __restrict__ out)
{
  __shared__ __bf16 Bs[2][32 * 136];
  const int t = threadIdx.x;
  const int wave = t >> 5, lane = t & 31;
  const int waveM = wave & 3, waveN = wave >> 2;
  const int m0 = blockIdx.x * 128;
  const int n0 = blockIdx.y * 128;

  v8f acc[2][4];
#pragma unroll
  for (int i = 0; i < 2; ++i)
#pragma unroll
    for (int j = 0; j < 4; ++j) acc[i][j] = v8f_zero();

  const int bcol = (t & 31) * 4, brow = t >> 5;

  float4 br[4];
  auto load_b = [&](int k0) {
#pragma unroll
    for (int rr = 0; rr < 4; ++rr)
      br[rr] = *(const float4*)(w + (size_t)(k0 + brow + rr * 8) * EMBED + n0 + bcol);
  };
  auto stage_b = [&](int buf) {
#pragma unroll
    for (int rr = 0; rr < 4; ++rr)
      store_bf16x4(&Bs[buf][(brow + rr * 8) * 136 + bcol], br[rr]);
  };

  load_b(0);
  stage_b(0);
  __syncthreads();

  const int NK = EMBED / 32;
  for (int kt = 0; kt < NK; ++kt) {
    const int cur = kt & 1;
    const int k0 = kt * 32;
    if (kt + 1 < NK) load_b((kt + 1) * 32);

    v16bf af[2], bfr[4];
#pragma unroll
    for (int i = 0; i < 2; ++i)
      af[i] = load_a_frag(ao + (size_t)(m0 + waveM * 32 + i * 16) * EMBED + k0,
                          EMBED, lane);
#pragma unroll
    for (int j = 0; j < 4; ++j)
      bfr[j] = load_b_frag(&Bs[cur][waveN * 64 + j * 16], 136, lane);
#pragma unroll
    for (int i = 0; i < 2; ++i)
#pragma unroll
      for (int j = 0; j < 4; ++j) acc[i][j] = wmma_bf16(af[i], bfr[j], acc[i][j]);

    if (kt + 1 < NK) stage_b(cur ^ 1);
    __syncthreads();
  }

  float bias_r[4];
#pragma unroll
  for (int j = 0; j < 4; ++j)
    bias_r[j] = bias[n0 + waveN * 64 + j * 16 + (lane & 15)];

#pragma unroll
  for (int i = 0; i < 2; ++i)
#pragma unroll
    for (int r = 0; r < 8; ++r) {
      const int mrow = m0 + waveM * 32 + i * 16 + ((lane < 16) ? r : (r + 8));
#pragma unroll
      for (int j = 0; j < 4; ++j) {
        const int ncol = n0 + waveN * 64 + j * 16 + (lane & 15);
        out[(size_t)mrow * EMBED + ncol] = acc[i][j][r] + bias_r[j];
      }
    }
}

// ---------------------------------------------------------------------------
extern "C" void kernel_launch(void* const* d_in, const int* in_sizes, int n_in,
                              void* d_out, int out_size, void* d_ws, size_t ws_size,
                              hipStream_t stream) {
  const float* x      = (const float*)d_in[0];
  const float* w_qkv  = (const float*)d_in[1];
  const float* b_qkv  = (const float*)d_in[2];
  const float* w_proj = (const float*)d_in[3];
  const float* b_proj = (const float*)d_in[4];
  float* out = (float*)d_out;

  // Workspace: 4 bf16 buffers of B*H*N*D = 4,194,304 elems (8 MB each, 32 MB total).
  const size_t perBuf = (size_t)BATCH * NH * SEQ * HD;
  __bf16* qb  = (__bf16*)d_ws;
  __bf16* ktb = qb  + perBuf;
  __bf16* vb  = ktb + perBuf;
  __bf16* ao  = vb  + perBuf;

  qkv_gemm_kernel<<<dim3(ROWS / 128, NQKV / 128), 256, 0, stream>>>(
      x, w_qkv, b_qkv, qb, ktb, vb);
  attn_kernel<<<dim3(BATCH * NH, SEQ / 128), 256, 0, stream>>>(qb, ktb, vb, ao);
  proj_gemm_kernel<<<dim3(ROWS / 128, EMBED / 128), 256, 0, stream>>>(
      ao, w_proj, b_proj, out);
}
